// Attention_40759239639448
// MI455X (gfx1250) — matrix operand order repaired and checked by reading, not compiled, not run on hardware
//
#include <hip/hip_runtime.h>
#include <hip/hip_bf16.h>

// ---------------------------------------------------------------------------
// CDNA5 (gfx1250) attention pipeline, bf16 WMMA everywhere.
//   stage 1: fp32 -> bf16 casts of x, w_qkv, w_out
//   stage 2: qkv = x @ w_qkv^T            (WMMA GEMM, f32 out)
//   stage 3: RoPE + RMSNorm, split q/k/v, v stored transposed
//   stage 4: causal flash attention       (WMMA QK^T and PV)
//   stage 5: out = attn @ w_out^T         (WMMA GEMM, f32 out)
// ---------------------------------------------------------------------------

typedef __attribute__((ext_vector_type(16))) __bf16 v16bf;
typedef __attribute__((ext_vector_type(8)))  float  v8f;
typedef __attribute__((ext_vector_type(4)))  float  v4f;

union FragBF {
    v16bf v;
    struct { v4f lo, hi; } f;
};

// A fragment, 16(M) x 32(K), from row-major [M][K]:
//   lanes 0-15 : row = m0+lane,    K = k0+0..7  (vgpr0-3) and k0+16..23 (vgpr4-7)
//   lanes 16-31: row = m0+lane-16, K = k0+8..15 and k0+24..31
static __device__ inline v16bf load_a_frag(const __hip_bfloat16* base, int ld,
                                           int row0, int k0, int lane) {
    const int r  = row0 + (lane & 15);
    const int kb = k0 + ((lane >> 4) << 3);
    const __hip_bfloat16* p = base + (size_t)r * ld + kb;
    FragBF u;
    u.f.lo = *(const v4f*)(p);
    u.f.hi = *(const v4f*)(p + 16);
    return u.v;
}

// B fragment, 32(K) x 16(N), loaded from row-major W[N][K] (i.e. B = W^T):
//   lanes 0-15 : col = n0+lane,    K = k0+0..7 and k0+16..23
//   lanes 16-31: col = n0+lane-16, K = k0+8..15 and k0+24..31
static __device__ inline v16bf load_b_frag(const __hip_bfloat16* base, int ld,
                                           int n0, int k0, int lane) {
    const int r  = n0 + (lane & 15);
    const int kb = k0 + ((lane >> 4) << 3);
    const __hip_bfloat16* p = base + (size_t)r * ld + kb;
    FragBF u;
    u.f.lo = *(const v4f*)(p);
    u.f.hi = *(const v4f*)(p + 16);
    return u.v;
}

#define WMMA_BF16(A, B, C) \
    __builtin_amdgcn_wmma_f32_16x16x32_bf16(false, (A), false, (B), (short)0, (C), false, false)

// ---------------------------------------------------------------------------
__global__ __launch_bounds__(256) void cvt_f32_to_bf16(const float* __restrict__ src,
                                                       __hip_bfloat16* __restrict__ dst,
                                                       int n) {
    int i = blockIdx.x * blockDim.x + threadIdx.x;
    if (i < n) dst[i] = __float2bfloat16(src[i]);
}

// ---------------------------------------------------------------------------
// C[M][N] (f32) = A[M][K] (bf16) @ B[N][K]^T (bf16)
// block: 256 threads = 8 waves, tile 64(M) x 128(N); wave tile 32x32 (2x2 wmma)
__global__ __launch_bounds__(256) void gemm_bf16_nt(const __hip_bfloat16* __restrict__ A,
                                                    const __hip_bfloat16* __restrict__ B,
                                                    float* __restrict__ C,
                                                    int M, int N, int K) {
    const int lane = threadIdx.x & 31;
    const int wave = threadIdx.x >> 5;
    const int m0 = blockIdx.y * 64 + (wave >> 2) * 32;
    const int n0 = blockIdx.x * 128 + (wave & 3) * 32;

    v8f c00 = {}; v8f c01 = {}; v8f c10 = {}; v8f c11 = {};

    for (int k = 0; k < K; k += 32) {
        v16bf a0 = load_a_frag(A, K, m0,      k, lane);
        v16bf a1 = load_a_frag(A, K, m0 + 16, k, lane);
        v16bf b0 = load_b_frag(B, K, n0,      k, lane);
        v16bf b1 = load_b_frag(B, K, n0 + 16, k, lane);
        c00 = WMMA_BF16(a0, b0, c00);
        c01 = WMMA_BF16(a0, b1, c01);
        c10 = WMMA_BF16(a1, b0, c10);
        c11 = WMMA_BF16(a1, b1, c11);
    }

    const int cn   = lane & 15;
    const int rofs = (lane >> 4) * 8;
#pragma unroll
    for (int r = 0; r < 8; ++r) {
        const int row = m0 + r + rofs;
        C[(size_t)row * N + n0 + cn]             = c00[r];
        C[(size_t)row * N + n0 + 16 + cn]        = c01[r];
        C[(size_t)(row + 16) * N + n0 + cn]      = c10[r];
        C[(size_t)(row + 16) * N + n0 + 16 + cn] = c11[r];
    }
}

// ---------------------------------------------------------------------------
// RoPE + RMSNorm on q,k; split qkv; v stored transposed [B,H,D,T].
// grid: (B*T, 2), 256 threads; wave w handles head h = blockIdx.y*8 + w.
__global__ __launch_bounds__(256) void rope_rms_split(const float* __restrict__ qkv,
                                                      const float* __restrict__ qw,
                                                      const float* __restrict__ kw,
                                                      __hip_bfloat16* __restrict__ qo,
                                                      __hip_bfloat16* __restrict__ ko,
                                                      __hip_bfloat16* __restrict__ vt,
                                                      int Tn, int Hn, int Dn) {
    const int lane = threadIdx.x & 31;
    const int wave = threadIdx.x >> 5;
    const int h   = blockIdx.y * 8 + wave;
    const int row = blockIdx.x;              // b*T + t
    const int b   = row / Tn;
    const int t   = row - b * Tn;
    const int Cn  = Hn * Dn;

    const float* qp = qkv + (size_t)row * (3 * Cn) + h * Dn;
    const float* kp = qp + Cn;
    const float* vp = qp + 2 * Cn;

    const float freq = __powf(10000.0f, -(float)lane / 32.0f);
    float sa, ca;
    __sincosf((float)t * freq, &sa, &ca);

    const size_t obase = ((size_t)(b * Hn + h) * Tn + t) * Dn;

    // --- q: rope then rmsnorm ---
    float x1 = qp[lane], x2 = qp[lane + 32];
    float r1 = x1 * ca - x2 * sa;
    float r2 = x2 * ca + x1 * sa;
    float ss = r1 * r1 + r2 * r2;
    for (int o = 16; o > 0; o >>= 1) ss += __shfl_xor(ss, o, 32);
    float inv = rsqrtf(ss * (1.0f / 64.0f) + 1e-6f);
    qo[obase + lane]      = __float2bfloat16(r1 * inv * qw[lane]);
    qo[obase + lane + 32] = __float2bfloat16(r2 * inv * qw[lane + 32]);

    // --- k: rope then rmsnorm ---
    x1 = kp[lane]; x2 = kp[lane + 32];
    r1 = x1 * ca - x2 * sa;
    r2 = x2 * ca + x1 * sa;
    ss = r1 * r1 + r2 * r2;
    for (int o = 16; o > 0; o >>= 1) ss += __shfl_xor(ss, o, 32);
    inv = rsqrtf(ss * (1.0f / 64.0f) + 1e-6f);
    ko[obase + lane]      = __float2bfloat16(r1 * inv * kw[lane]);
    ko[obase + lane + 32] = __float2bfloat16(r2 * inv * kw[lane + 32]);

    // --- v: transpose into [B,H,D,T] ---
    const size_t vb = ((size_t)(b * Hn + h) * Dn) * Tn + t;
    vt[vb + (size_t)lane * Tn]        = __float2bfloat16(vp[lane]);
    vt[vb + (size_t)(lane + 32) * Tn] = __float2bfloat16(vp[lane + 32]);
}

// ---------------------------------------------------------------------------
// Causal flash attention. One wave owns a 16-row q tile, streams 32-key blocks.
// grid: (T/128, B*H), 256 threads (8 waves).
__global__ __launch_bounds__(256) void flash_attn(const __hip_bfloat16* __restrict__ Q,
                                                  const __hip_bfloat16* __restrict__ Km,
                                                  const __hip_bfloat16* __restrict__ Vt,
                                                  __hip_bfloat16* __restrict__ Ao,
                                                  int Tn, int Hn, int Dn) {
    __shared__ __hip_bfloat16 plds[8][16 * 32];   // per-wave P staging (C-layout -> A-layout)

    const int lane = threadIdx.x & 31;
    const int wave = threadIdx.x >> 5;
    const int bh   = blockIdx.y;
    const int b    = bh / Hn;
    const int h    = bh - b * Hn;
    const int q0   = blockIdx.x * 128 + wave * 16;
    const int Cn   = Hn * Dn;

    const __hip_bfloat16* qp = Q  + (size_t)bh * Tn * Dn;
    const __hip_bfloat16* kp = Km + (size_t)bh * Tn * Dn;
    const __hip_bfloat16* vp = Vt + (size_t)bh * Dn * Tn;

    const v16bf aq0 = load_a_frag(qp, Dn, q0, 0,  lane);
    const v16bf aq1 = load_a_frag(qp, Dn, q0, 32, lane);

    v8f o0 = {}; v8f o1 = {}; v8f o2 = {}; v8f o3 = {};
    float mrow[8], lrow[8];
#pragma unroll
    for (int r = 0; r < 8; ++r) { mrow[r] = -1e30f; lrow[r] = 0.0f; }

    const int   half  = lane >> 4;
    const int   cn    = lane & 15;
    const float scale = 0.125f;               // 1/sqrt(64)

    for (int kb = 0; kb < q0 + 16; kb += 32) {
        // ---- S = Q K^T (two 16x16 score tiles, K-dim = D = 64) ----
        v16bf bk00 = load_b_frag(kp, Dn, kb,      0,  lane);
        v16bf bk01 = load_b_frag(kp, Dn, kb,      32, lane);
        v16bf bk10 = load_b_frag(kp, Dn, kb + 16, 0,  lane);
        v16bf bk11 = load_b_frag(kp, Dn, kb + 16, 32, lane);

        v8f s0 = {}; v8f s1 = {};
        s0 = WMMA_BF16(aq0, bk00, s0);
        s0 = WMMA_BF16(aq1, bk01, s0);
        s1 = WMMA_BF16(aq0, bk10, s1);
        s1 = WMMA_BF16(aq1, bk11, s1);

        // ---- causal mask + online softmax (row = M mapping of C-layout) ----
#pragma unroll
        for (int r = 0; r < 8; ++r) {
            const int   qrow = q0 + r + half * 8;
            const int   key0 = kb + cn;
            const int   key1 = key0 + 16;
            float v0 = (key0 <= qrow) ? s0[r] * scale : -1e30f;
            float v1 = (key1 <= qrow) ? s1[r] * scale : -1e30f;

            float mx = fmaxf(v0, v1);
            for (int o = 8; o > 0; o >>= 1) mx = fmaxf(mx, __shfl_xor(mx, o, 32));
            const float mnew  = fmaxf(mrow[r], mx);
            const float alpha = __expf(mrow[r] - mnew);
            const float p0 = __expf(v0 - mnew);
            const float p1 = __expf(v1 - mnew);
            float psum = p0 + p1;
            for (int o = 8; o > 0; o >>= 1) psum += __shfl_xor(psum, o, 32);

            lrow[r] = lrow[r] * alpha + psum;
            mrow[r] = mnew;
            o0[r] *= alpha; o1[r] *= alpha; o2[r] *= alpha; o3[r] *= alpha;

            plds[wave][(r + half * 8) * 32 + cn]      = __float2bfloat16(p0);
            plds[wave][(r + half * 8) * 32 + cn + 16] = __float2bfloat16(p1);
        }
        asm volatile("s_wait_dscnt 0" ::: "memory");   // DS in-order per wave; explicit anyway

        // ---- re-load P in A-fragment layout from LDS ----
        const __hip_bfloat16* lp = &plds[wave][cn * 32 + half * 8];
        FragBF pu;
        pu.f.lo = *(const v4f*)(lp);
        pu.f.hi = *(const v4f*)(lp + 16);
        const v16bf pf = pu.v;

        // ---- O += P @ V  (V^T stored [D][T], 4 column tiles of D=64) ----
        v16bf bv0 = load_b_frag(vp, Tn, 0,  kb, lane);
        v16bf bv1 = load_b_frag(vp, Tn, 16, kb, lane);
        v16bf bv2 = load_b_frag(vp, Tn, 32, kb, lane);
        v16bf bv3 = load_b_frag(vp, Tn, 48, kb, lane);
        o0 = WMMA_BF16(pf, bv0, o0);
        o1 = WMMA_BF16(pf, bv1, o1);
        o2 = WMMA_BF16(pf, bv2, o2);
        o3 = WMMA_BF16(pf, bv3, o3);
    }

    // ---- normalize and store to [B,T,H*D] as bf16 for the output GEMM ----
#pragma unroll
    for (int r = 0; r < 8; ++r) {
        const int   tq   = q0 + r + half * 8;
        const float invl = 1.0f / lrow[r];
        const size_t base = ((size_t)(b * Tn + tq)) * Cn + h * Dn;
        Ao[base + cn]      = __float2bfloat16(o0[r] * invl);
        Ao[base + 16 + cn] = __float2bfloat16(o1[r] * invl);
        Ao[base + 32 + cn] = __float2bfloat16(o2[r] * invl);
        Ao[base + 48 + cn] = __float2bfloat16(o3[r] * invl);
    }
}

// ---------------------------------------------------------------------------
extern "C" void kernel_launch(void* const* d_in, const int* in_sizes, int n_in,
                              void* d_out, int out_size, void* d_ws, size_t ws_size,
                              hipStream_t stream) {
    (void)in_sizes; (void)n_in; (void)out_size; (void)ws_size;

    const int Bn = 2, Tn = 2048, Cn = 1024, Hn = 16, Dn = 64;
    const int Mn = Bn * Tn;          // 4096 tokens
    const int Nqkv = 3 * Cn;         // 3072

    const float* x     = (const float*)d_in[0];
    const float* w_qkv = (const float*)d_in[1];
    const float* w_out = (const float*)d_in[2];
    const float* qnw   = (const float*)d_in[3];
    const float* knw   = (const float*)d_in[4];
    float* out = (float*)d_out;

    // workspace layout (bytes)
    char* ws = (char*)d_ws;
    __hip_bfloat16* xb   = (__hip_bfloat16*)ws;  ws += (size_t)Mn * Cn * 2;     //  8 MiB
    __hip_bfloat16* wqb  = (__hip_bfloat16*)ws;  ws += (size_t)Nqkv * Cn * 2;   //  6 MiB
    __hip_bfloat16* wob  = (__hip_bfloat16*)ws;  ws += (size_t)Cn * Cn * 2;     //  2 MiB
    float*          qkvf = (float*)ws;           ws += (size_t)Mn * Nqkv * 4;   // 48 MiB
    __hip_bfloat16* qb   = (__hip_bfloat16*)ws;  ws += (size_t)Mn * Cn * 2;     //  8 MiB
    __hip_bfloat16* kbuf = (__hip_bfloat16*)ws;  ws += (size_t)Mn * Cn * 2;     //  8 MiB
    __hip_bfloat16* vtb  = (__hip_bfloat16*)ws;  ws += (size_t)Mn * Cn * 2;     //  8 MiB
    __hip_bfloat16* attb = (__hip_bfloat16*)ws;  ws += (size_t)Mn * Cn * 2;     //  8 MiB

    // stage 1: casts
    {
        int n = Mn * Cn;
        cvt_f32_to_bf16<<<(n + 255) / 256, 256, 0, stream>>>(x, xb, n);
        n = Nqkv * Cn;
        cvt_f32_to_bf16<<<(n + 255) / 256, 256, 0, stream>>>(w_qkv, wqb, n);
        n = Cn * Cn;
        cvt_f32_to_bf16<<<(n + 255) / 256, 256, 0, stream>>>(w_out, wob, n);
    }

    // stage 2: qkv = x @ w_qkv^T   (4096 x 3072 x 1024)
    gemm_bf16_nt<<<dim3(Nqkv / 128, Mn / 64), 256, 0, stream>>>(xb, wqb, qkvf, Mn, Nqkv, Cn);

    // stage 3: RoPE + RMSNorm, split, transpose V
    rope_rms_split<<<dim3(Mn, 2), 256, 0, stream>>>(qkvf, qnw, knw, qb, kbuf, vtb, Tn, Hn, Dn);

    // stage 4: causal flash attention
    flash_attn<<<dim3(Tn / 128, Bn * Hn), 256, 0, stream>>>(qb, kbuf, vtb, attb, Tn, Hn, Dn);

    // stage 5: out = attn @ w_out^T   (4096 x 1024 x 1024)
    gemm_bf16_nt<<<dim3(Cn / 128, Mn / 64), 256, 0, stream>>>(attb, wob, out, Mn, Cn, Cn);
}
